// NodeEdge_64458869178762
// MI455X (gfx1250) — compile-verified, matrix operand order checked by reading, not run on hardware
//
#include <hip/hip_runtime.h>

typedef __attribute__((ext_vector_type(16))) _Float16 v16h;
typedef __attribute__((ext_vector_type(8)))  float    v8f;

namespace {
constexpr int kNB = 16;        // batches
constexpr int kNN = 8192;      // nodes
constexpr int kNE = 262144;    // edges
constexpr int kND = 8;         // feature dim
constexpr int kWPR = kNN / 32; // 256 bitmask words per adjacency row
constexpr int kKB  = kNN / 32; // 256 K-blocks (K-step 32 per WMMA)
constexpr int kCT  = (kNB * kND) / 16; // 8 column tiles (128 packed cols)
constexpr int kKSplit = 8;             // waves per workgroup splitting K
constexpr int kItersPerWave = kKB / kKSplit; // 32
constexpr int kPairs = kItersPerWave / 2;    // 16 uint2 bit-word pairs per wave

// workspace layout (bytes)
constexpr size_t kOffBits   = 0;                                   // 8 MiB bitmask
constexpr size_t kOffInvDeg = (size_t)kNN * kWPR * 4;              // 32 KiB
constexpr size_t kOffSelf   = kOffInvDeg + (size_t)kNN * 4;        // 32 KiB
constexpr size_t kOffBpack  = kOffSelf  + (size_t)kNN * 4;         // 2 MiB f16, 32B aligned
}

// ---------------------------------------------------------------------------
// 1) bitmask init: zero all words, pre-set diagonal bit of each row,
//    zero the self-edge flags. No atomics needed (one writer per word).
__global__ void k_init_mask(unsigned* __restrict__ bits, float* __restrict__ selfw) {
  int idx = blockIdx.x * blockDim.x + threadIdx.x;       // [0, 2M)
  int row  = idx >> 8;                                   // / 256 words per row
  int wcol = idx & 255;
  bits[idx] = (wcol == (row >> 5)) ? (1u << (row & 31)) : 0u;
  if (idx < kNN) selfw[idx] = 0.0f;
}

// 2) symmetric set-scatter of ALL B*E edges into the bitmask (set semantics
//    dedups duplicates exactly like jnp .at[].set(1.0)). Self edges flagged:
//    reference does set(1.0) then +eye, so those diagonals equal 2.0.
__global__ void k_scatter(const int* __restrict__ ai, const int* __restrict__ aj,
                          unsigned* __restrict__ bits, float* __restrict__ selfw) {
  int t = blockIdx.x * blockDim.x + threadIdx.x;         // [0, B*E)
  int i = ai[t] & (kNN - 1);
  int j = aj[t] & (kNN - 1);
  atomicOr(bits + (size_t)i * kWPR + (j >> 5), 1u << (j & 31));
  atomicOr(bits + (size_t)j * kWPR + (i >> 5), 1u << (i & 31));
  if (i == j) selfw[i] = 1.0f;                           // benign race, all write 1.0
}

// 3) row sums: popcount of row bits (+1 for nodes with an explicit self edge).
__global__ void k_degree(const unsigned* __restrict__ bits,
                         const float* __restrict__ selfw,
                         float* __restrict__ inv_deg) {
  int row = blockIdx.x * blockDim.x + threadIdx.x;
  if (row >= kNN) return;
  const uint4* p = (const uint4*)(bits + (size_t)row * kWPR);
  int s = 0;
#pragma unroll 8
  for (int w = 0; w < kWPR / 4; ++w) {
    uint4 q = p[w];
    s += __popc(q.x) + __popc(q.y) + __popc(q.z) + __popc(q.w);
  }
  inv_deg[row] = 1.0f / ((float)s + selfw[row]);
}

// 4) pack h_nodes (B,N,D) f32 -> f16 B-operand tiles in exactly the VGPR
//    layout v_wmma_f32_16x16x32_f16 expects for its B matrix (32x16, K x N):
//    lane L holds col N = (L&15); half h holds K = (h<8 ? h : h+8) + (L>=16 ? 8:0).
//    Packed col n = b*8 + d  (one 8192x128 GEMM covers all batches).
//    Tile (kb,c) for lane L lives at 32 contiguous bytes -> single v16h load.
__global__ void k_pack(const float* __restrict__ h_nodes, _Float16* __restrict__ bpack) {
  int t = blockIdx.x * blockDim.x + threadIdx.x;         // [0, 524288) dwords
  int idx = t * 2;                                       // half index
  int h  = idx & 15;
  int L  = (idx >> 4) & 31;
  int c  = (idx >> 9) & 7;
  int kb = idx >> 12;
  int base = (L & 16) ? 8 : 0;
  int n = c * 16 + (L & 15);
  int b = n >> 3, d = n & 7;
#pragma unroll
  for (int q = 0; q < 2; ++q) {
    int hh = h + q;
    int k = kb * 32 + ((hh < 8) ? hh : hh + 8) + base;
    float v = h_nodes[((size_t)b * kNN + k) * kND + d];
    bpack[idx + q] = (_Float16)v;
  }
}

// 5) edge_out = (h[ai]+h[aj]) @ W.T + bias  -- bandwidth bound (128 MB write).
//    W/bias are wave-uniform (scalar cached); h gathers hit L2 (h is 4 MB).
__global__ void __launch_bounds__(256)
k_edge(const float* __restrict__ h_nodes, const int* __restrict__ ai,
       const int* __restrict__ aj, const float* __restrict__ W,
       const float* __restrict__ bias, float* __restrict__ out) {
  int e = blockIdx.x * blockDim.x + threadIdx.x;         // [0, E)
  float w[64], bb[8];
#pragma unroll
  for (int t = 0; t < 64; ++t) w[t] = W[t];              // uniform -> SGPRs
#pragma unroll
  for (int d = 0; d < 8; ++d) bb[d] = bias[d];
  int i = ai[e] & (kNN - 1);
  int j = aj[e] & (kNN - 1);
  for (int b = 0; b < kNB; ++b) {
    const float4* p1 = (const float4*)(h_nodes + ((size_t)b * kNN + i) * kND);
    const float4* p2 = (const float4*)(h_nodes + ((size_t)b * kNN + j) * kND);
    float4 a0 = p1[0], a1 = p1[1], c0 = p2[0], c1 = p2[1];
    float s[8] = {a0.x + c0.x, a0.y + c0.y, a0.z + c0.z, a0.w + c0.w,
                  a1.x + c1.x, a1.y + c1.y, a1.z + c1.z, a1.w + c1.w};
    float o[8];
#pragma unroll
    for (int d = 0; d < 8; ++d) {
      float acc = bb[d];
#pragma unroll
      for (int k = 0; k < 8; ++k) acc = fmaf(s[k], w[d * 8 + k], acc);
      o[d] = acc;
    }
    float4* po = (float4*)(out + ((size_t)b * kNE + e) * kND);
    po[0] = make_float4(o[0], o[1], o[2], o[3]);
    po[1] = make_float4(o[4], o[5], o[6], o[7]);
  }
}

// ---------------------------------------------------------------------------
// A-tile synthesis: one bitmask dword -> 16 f16 {0,1} halves in the ISA's
// 16-bit A-matrix 16x32 layout (halves 0..7: K=base..base+7, 8..15: K=base+16..23).
__device__ __forceinline__ v16h make_a(unsigned wbits, int base) {
  unsigned lo = (wbits >> base) & 0xFFu;
  unsigned hi = (wbits >> (base + 16)) & 0xFFu;
  const _Float16 one = (_Float16)1.0f;
  const _Float16 zer = (_Float16)0.0f;
  v16h a;
#pragma unroll
  for (int h = 0; h < 8; ++h) a[h] = ((lo >> h) & 1u) ? one : zer;
#pragma unroll
  for (int h = 0; h < 8; ++h) a[8 + h] = ((hi >> h) & 1u) ? one : zer;
  return a;
}

// 6) new_h = rownorm(adj) @ h  as dense f16 WMMA over the bitmask.
//    Workgroup = 256 threads = 8 waves, all owning the SAME 16-row strip but
//    each a 1/8 slice of K (32 K-blocks) -> 4096 waves chip-wide.
//    Steady-state loop has NO wait-to-zero: B tiles double-buffered in
//    registers AND the adjacency bit-words pipelined one iteration ahead,
//    so only counted s_wait_loadcnt remain. Partials reduced through LDS;
//    wave w does the scale/self-edge epilogue for column tile c = w.
__global__ void __launch_bounds__(256)
k_spmm(const unsigned* __restrict__ bits, const _Float16* __restrict__ bpack,
       const float* __restrict__ inv_deg, const float* __restrict__ selfw,
       const float* __restrict__ h_nodes, float* __restrict__ out_nodes) {
  __shared__ float red[kKSplit][kCT][32][8];             // 64 KiB partials
  int lane = threadIdx.x & 31;
  int wave = threadIdx.x >> 5;                           // K-split index [0,8)
  int r = blockIdx.x;                                    // row block [0, 512)
  int m = lane & 15;                                     // local row (A) / col (B)
  int row = r * 16 + m;
  int base = (lane & 16) ? 8 : 0;                        // K sub-offset per lane half

  v8f zero = {0.f, 0.f, 0.f, 0.f, 0.f, 0.f, 0.f, 0.f};
  v8f acc[kCT];
#pragma unroll
  for (int c = 0; c < kCT; ++c) acc[c] = zero;

  int kb0 = wave * kItersPerWave;                        // this wave's K-block range
  const uint2* rb2 = (const uint2*)(bits + (size_t)row * kWPR + kb0); // 16 word pairs
  const v16h* bp = (const v16h*)bpack;                   // tile stride: 8 ctiles * 32 lanes

  // register double-buffered B tiles (b0 even steps, b1 odd) + pipelined bits
  v16h b0[kCT], b1[kCT];
  uint2 wpair = rb2[0];
  {
    size_t tb = (size_t)kb0 * (kCT * 32);
#pragma unroll
    for (int c = 0; c < kCT; ++c) b0[c] = bp[tb + (size_t)c * 32 + lane];
  }
  for (int it2 = 0; it2 < kPairs; ++it2) {
    int it = it2 * 2;
    // fetch NEXT pair of adjacency words early (branchless wrap on last iter);
    // consumed only at the bottom, so no wait-to-zero in the steady state.
    uint2 wnext = rb2[(it2 + 1) & (kPairs - 1)];

    size_t tb1 = (size_t)(kb0 + it + 1) * (kCT * 32);
#pragma unroll
    for (int c = 0; c < kCT; ++c) b1[c] = bp[tb1 + (size_t)c * 32 + lane];
    v16h a0 = make_a(wpair.x, base);
#pragma unroll
    for (int c = 0; c < kCT; ++c)
      acc[c] = __builtin_amdgcn_wmma_f32_16x16x32_f16(
          false, a0, false, b0[c], (short)0, acc[c], false, false);

    if (it + 2 < kItersPerWave) {
      size_t tb2 = (size_t)(kb0 + it + 2) * (kCT * 32);
#pragma unroll
      for (int c = 0; c < kCT; ++c) b0[c] = bp[tb2 + (size_t)c * 32 + lane];
    }
    v16h a1 = make_a(wpair.y, base);
#pragma unroll
    for (int c = 0; c < kCT; ++c)
      acc[c] = __builtin_amdgcn_wmma_f32_16x16x32_f16(
          false, a1, false, b1[c], (short)0, acc[c], false, false);

    wpair = wnext;
  }

  // cross-wave K reduction through LDS
#pragma unroll
  for (int c = 0; c < kCT; ++c)
    *(v8f*)&red[wave][c][lane][0] = acc[c];
  __syncthreads();

  // wave `wave` owns column tile c = wave for the epilogue.
  int c = wave;
  v8f tot = zero;
#pragma unroll
  for (int p = 0; p < kKSplit; ++p)
    tot += *(const v8f*)&red[p][c][lane][0];

  // C/D layout: tot[v] = OUT[row = r*16+v+(lane>=16?8:0), col = c*16+(lane&15)];
  // packed col n -> (b = n>>3, d = n&7). Add self-edge term, scale by 1/rowsum.
#pragma unroll
  for (int v = 0; v < 8; ++v) {
    int rr = r * 16 + v + ((lane & 16) ? 8 : 0);
    float invd = inv_deg[rr];
    float sf = selfw[rr];
    int n = c * 16 + m;
    int b = n >> 3, d = n & 7;
    size_t off = ((size_t)b * kNN + rr) * kND + d;
    float hv = h_nodes[off];
    out_nodes[off] = (tot[v] + sf * hv) * invd;
  }
}

// ---------------------------------------------------------------------------
extern "C" void kernel_launch(void* const* d_in, const int* in_sizes, int n_in,
                              void* d_out, int out_size, void* d_ws, size_t ws_size,
                              hipStream_t stream) {
  const float* h_nodes = (const float*)d_in[0];
  const int*   adj_i   = (const int*)d_in[1];   // full (B,E) for adjacency
  const int*   adj_j   = (const int*)d_in[2];
  const float* W       = (const float*)d_in[3];
  const float* bias    = (const float*)d_in[4];
  float* out = (float*)d_out;

  unsigned char* ws = (unsigned char*)d_ws;
  unsigned*  bits    = (unsigned*)(ws + kOffBits);
  float*     inv_deg = (float*)(ws + kOffInvDeg);
  float*     selfw   = (float*)(ws + kOffSelf);
  _Float16*  bpack   = (_Float16*)(ws + kOffBpack);

  // 1) init bitmask (+diag) & self flags
  k_init_mask<<<(kNN * kWPR) / 256, 256, 0, stream>>>(bits, selfw);
  // 2) symmetric set-scatter of all B*E edges
  k_scatter<<<(kNB * kNE) / 256, 256, 0, stream>>>(adj_i, adj_j, bits, selfw);
  // 3) row sums -> 1/deg
  k_degree<<<kNN / 256, 256, 0, stream>>>(bits, selfw, inv_deg);
  // 4) pack h into WMMA B-operand tiles (f16)
  k_pack<<<(kNN * kNB * kND / 2) / 256, 256, 0, stream>>>(h_nodes, bpack);
  // 5) edge output (first 128 MB of d_out)
  k_edge<<<kNE / 256, 256, 0, stream>>>(h_nodes, adj_i, adj_j, W, bias, out);
  // 6) normalized adjacency matmul via WMMA (last 4 MB of d_out)
  k_spmm<<<512, 256, 0, stream>>>(bits, bpack, inv_deg, selfw, h_nodes,
                                  out + (size_t)kNB * kNE * kND);
}
